// MultiHeadAttention_69664369541885
// MI455X (gfx1250) — compile-verified, matrix operand order checked by reading, not run on hardware
//
#include <hip/hip_runtime.h>
#include <hip/hip_bf16.h>

// Problem constants (match reference)
#define BATCH 8
#define SEQ   1024
#define HID_  1024
#define NHEAD 16
#define HDIM  64

typedef __attribute__((ext_vector_type(16))) __bf16 v16bf;
typedef __attribute__((ext_vector_type(8)))  float  v8f;

typedef unsigned int u32x4 __attribute__((ext_vector_type(4)));
typedef int          i32x4 __attribute__((ext_vector_type(4)));
typedef int          i32x8 __attribute__((ext_vector_type(8)));

#if defined(__AMDGCN__) && __has_builtin(__builtin_amdgcn_tensor_load_to_lds)
#define HAVE_TDM 1
#else
#define HAVE_TDM 0
#endif

// CDNA5 WMMA 16x16x32 bf16 fragment k-index mapping (ISA 7.12.2):
// j in [0,16): k = (j/8)*16 + ((lane>=16)?8:0) + (j%8)
__device__ __forceinline__ int kmap(int j, int lane) {
    return ((j >> 3) << 4) + ((lane >> 4) << 3) + (j & 7);
}

#if HAVE_TDM
// Issue a 2D TDM tile load (bf16 elements) global -> LDS.
// D# layout per CDNA5 ISA:
//  group0: [1:0]=count=1, [63:32]=lds_addr, [120:64]=global_addr, [127:126]=type=2
//  group1: [17:16]=data_size(1 => 2B), [79:48]=tensor_dim0, [111:80]=tensor_dim1,
//          [127:112]=tile_dim0, [143:128]=tile_dim1, [207:160]=tensor_dim0_stride
__device__ __forceinline__ void tdm_load_2d_bf16(unsigned lds_off, const void* gptr,
                                                 unsigned tensor_d0, unsigned tensor_d1,
                                                 unsigned stride0,
                                                 unsigned tile_d0, unsigned tile_d1) {
    unsigned long long ga = (unsigned long long)gptr;
    u32x4 g0;
    g0[0] = 1u;                                                // count=1 (valid user D#)
    g0[1] = lds_off;                                           // lds_addr (bytes)
    g0[2] = (unsigned)ga;                                      // global_addr[31:0]
    g0[3] = (unsigned)((ga >> 32) & 0x1FFFFFFu) | (2u << 30);  // addr[56:32] | type=2
    i32x8 g1;
    g1[0] = 0x00010000;                                        // data_size=1 (2 bytes)
    g1[1] = (int)((tensor_d0 & 0xFFFFu) << 16);                // tensor_dim0[15:0]
    g1[2] = (int)((tensor_d0 >> 16) | ((tensor_d1 & 0xFFFFu) << 16));
    g1[3] = (int)((tensor_d1 >> 16) | (tile_d0 << 16));        // tile_dim0
    g1[4] = (int)tile_d1;                                      // tile_dim1 (tile_dim2=0)
    g1[5] = (int)stride0;                                      // tensor_dim0_stride[31:0]
    g1[6] = 0;
    g1[7] = 0;
    i32x4 z4 = {0, 0, 0, 0};
#if __clang_major__ >= 23
    i32x8 z8 = {0, 0, 0, 0, 0, 0, 0, 0};
    __builtin_amdgcn_tensor_load_to_lds(g0, g1, z4, z4, z8, 0);
#else
    __builtin_amdgcn_tensor_load_to_lds(g0, g1, z4, z4, 0);
#endif
}
#endif  // HAVE_TDM

// ---------------------------------------------------------------------------
// Kernel 0: f32 -> bf16 elementwise conversion
// ---------------------------------------------------------------------------
__global__ void mha_cvt_f32_bf16(const float* __restrict__ in,
                                 __bf16* __restrict__ out, int n) {
    int i = blockIdx.x * blockDim.x + threadIdx.x;
    if (i < n) out[i] = (__bf16)in[i];
}

// ---------------------------------------------------------------------------
// Kernel 1: bf16 WMMA GEMM with bias:  out[M,N] = X[M,K] * W[K,N] + bias[N]
// Block tile: 16(M) x 128(N), 8 waves (256 thr), 64-wide K steps.
// A (16x64, 2KB) and B (64x128, 16KB) tiles staged in LDS by the Tensor Data
// Mover (wave0 -> B tile, wave1 -> A tile, s_wait_tensorcnt, barrier); falls
// back to synchronous dword staging if the TDM builtin is unavailable.
// Requires M%16==0, N%128==0, K%64==0.
// ---------------------------------------------------------------------------
template <bool OUT_F32>
__global__ void mha_gemm_bf16_wmma(const __bf16* __restrict__ X,
                                   const __bf16* __restrict__ W,
                                   const float*  __restrict__ bias,
                                   float*  __restrict__ outF,
                                   __bf16* __restrict__ outB,
                                   int M, int N, int K) {
    __shared__ __align__(16) __bf16 As[16 * 64];    // 2 KB
    __shared__ __align__(16) __bf16 Bs[64 * 128];   // 16 KB
    const int tid  = threadIdx.x;
    const int wave = tid >> 5;
    const int lane = tid & 31;
    const int m0 = blockIdx.y * 16;
    const int n0 = blockIdx.x * 128;

    v8f acc = {};

    for (int k0 = 0; k0 < K; k0 += 64) {
#if HAVE_TDM
        if (wave == 0) {
            // B tile: rows k0..k0+63, cols n0..n0+127 of W[K,N]
            tdm_load_2d_bf16((unsigned)(size_t)(__bf16*)Bs,
                             W + (size_t)k0 * N + n0,
                             (unsigned)N, (unsigned)K, (unsigned)N, 128u, 64u);
            __builtin_amdgcn_s_wait_tensorcnt(0);
        } else if (wave == 1) {
            // A tile: rows m0..m0+15, cols k0..k0+63 of X[M,K]
            tdm_load_2d_bf16((unsigned)(size_t)(__bf16*)As,
                             X + (size_t)m0 * K + k0,
                             (unsigned)K, (unsigned)M, (unsigned)K, 64u, 16u);
            __builtin_amdgcn_s_wait_tensorcnt(0);
        }
#else
        {   // ---- stage A tile 16x64 (512 dwords, 2 per thread) ----
            unsigned int* dstA = (unsigned int*)As;
#pragma unroll
            for (int i = 0; i < 2; ++i) {
                const int di  = tid + (i << 8);
                const int row = di >> 5;
                const int cp  = di & 31;
                dstA[di] = *(const unsigned int*)(X + (size_t)(m0 + row) * K +
                                                  k0 + (cp << 1));
            }
        }
        {   // ---- stage B tile 64x128 (4096 dwords, 16 per thread) ----
            unsigned int* dstB = (unsigned int*)Bs;
#pragma unroll
            for (int i = 0; i < 16; ++i) {
                const int di  = tid + (i << 8);
                const int row = di >> 6;
                const int cp  = di & 63;
                dstB[di] = *(const unsigned int*)(W + (size_t)(k0 + row) * N +
                                                  n0 + (cp << 1));
            }
        }
#endif
        // Prefetch next A K-tile into caches (gfx1250 global_prefetch_b8)
        if (k0 + 64 < K)
            __builtin_prefetch(X + (size_t)(m0 + (tid >> 4)) * K + k0 + 64, 0, 0);
        __syncthreads();

        // ---- two 32-wide WMMA steps from LDS ----
#pragma unroll
        for (int ks2 = 0; ks2 < 2; ++ks2) {
            v16bf a, b;
#pragma unroll
            for (int j = 0; j < 16; ++j) {
                const int kk = (ks2 << 5) + kmap(j, lane);
                a[j] = As[(lane & 15) * 64 + kk];
                b[j] = Bs[kk * 128 + (wave << 4) + (lane & 15)];
            }
            acc = __builtin_amdgcn_wmma_f32_16x16x32_bf16(
                false, a, false, b, (short)0, acc, false, false);
        }
        __syncthreads();
    }

    // ---- epilogue: bias add + store ----
    const int n  = n0 + (wave << 4) + (lane & 15);
    const float bn = bias[n];
#pragma unroll
    for (int r = 0; r < 8; ++r) {
        const int m = m0 + r + ((lane >> 4) << 3);
        const float v = acc[r] + bn;
        if (OUT_F32) outF[(size_t)m * N + n] = v;
        else         outB[(size_t)m * N + n] = (__bf16)v;
    }
}

// ---------------------------------------------------------------------------
// Kernel 2: scores + softmax.
// One block per (b, h, 16-row query tile); 128 threads (4 waves).
// S[16][1024] strip kept in LDS (64 KB, fine in CDNA5's 320KB/WGP),
// softmax over k, f32 weights written once to d_out.
// ---------------------------------------------------------------------------
__global__ void mha_scores_softmax(const __bf16* __restrict__ Qp,   // [B*T, HID]
                                   const __bf16* __restrict__ Kp,   // [B*T, HID]
                                   float* __restrict__ attnW) {     // [B,NH,T,T]
    __shared__ __align__(16) float S[16 * SEQ];   // 64 KB
    __shared__ float rmax[16];
    __shared__ float rinv[16];

    const int blk = blockIdx.x;
    const int qt = blk & 63;          // T/16 = 64 tiles
    const int h  = (blk >> 6) & 15;
    const int b  = blk >> 10;
    const int q0 = qt << 4;
    const int tid  = threadIdx.x;
    const int wave = tid >> 5;
    const int lane = tid & 31;

    // Q fragments for this 16-row tile (two 32-wide K steps over d=0..63)
    v16bf aq[2];
    {
        const __bf16* qrow =
            Qp + ((size_t)(b * SEQ + q0 + (lane & 15))) * HID_ + h * HDIM;
#pragma unroll
        for (int ks = 0; ks < 2; ++ks)
#pragma unroll
            for (int j = 0; j < 16; ++j)
                aq[ks][j] = qrow[(ks << 5) + kmap(j, lane)];
    }

    const float scale = 0.125f;   // 1/sqrt(HD)

    // Each wave covers 16 of the 64 key tiles
    for (int kt = wave; kt < 64; kt += 4) {
        v8f acc = {};
#pragma unroll
        for (int ks = 0; ks < 2; ++ks) {
            v16bf bk;
            const __bf16* base =
                Kp + ((size_t)(b * SEQ + (kt << 4) + (lane & 15))) * HID_ +
                h * HDIM + (ks << 5);
#pragma unroll
            for (int j = 0; j < 16; ++j) bk[j] = base[kmap(j, lane)];
            acc = __builtin_amdgcn_wmma_f32_16x16x32_bf16(
                false, aq[ks], false, bk, (short)0, acc, false, false);
        }
        const int c = (kt << 4) + (lane & 15);
#pragma unroll
        for (int r = 0; r < 8; ++r) {
            const int m = r + ((lane >> 4) << 3);
            S[m * SEQ + c] = acc[r] * scale;
        }
    }
    __syncthreads();

    // Row softmax statistics (one thread per row)
    if (tid < 16) {
        float mx = -3.402823466e38f;
        for (int c = 0; c < SEQ; ++c) mx = fmaxf(mx, S[tid * SEQ + c]);
        float sm = 0.f;
        for (int c = 0; c < SEQ; ++c) sm += __expf(S[tid * SEQ + c] - mx);
        rmax[tid] = mx;
        rinv[tid] = 1.f / sm;
    }
    __syncthreads();

    // Normalize + single f32 write of the weights (HBM-bound part)
    float* dst = attnW + (((size_t)(b * NHEAD + h)) * SEQ + q0) * SEQ;
    for (int idx = tid; idx < 16 * SEQ; idx += 128) {
        const int r = idx >> 10;
        const int c = idx & (SEQ - 1);
        dst[(size_t)r * SEQ + c] = __expf(S[idx] - rmax[r]) * rinv[r];
    }
}

// ---------------------------------------------------------------------------
// Kernel 3: context = attnW @ V per head. One wave per (b,h,16-row q tile);
// 4 accumulators cover the full 16x64 head output. A built on the fly from
// the f32 weights (->bf16), B from bf16 V.
// ---------------------------------------------------------------------------
__global__ void mha_context(const float* __restrict__ attnW,  // [B,NH,T,T]
                            const __bf16* __restrict__ Vp,    // [B*T, HID]
                            __bf16* __restrict__ ctx) {       // [B*T, HID]
    const int blk = blockIdx.x;
    const int qt = blk & 63;
    const int h  = (blk >> 6) & 15;
    const int b  = blk >> 10;
    const int q0 = qt << 4;
    const int lane = threadIdx.x & 31;

    v8f acc[4] = {};
    const float* wrow =
        attnW + (((size_t)(b * NHEAD + h)) * SEQ + q0 + (lane & 15)) * (size_t)SEQ;

    for (int ks = 0; ks < 32; ++ks) {         // 32 key positions per step
        v16bf a;
#pragma unroll
        for (int j = 0; j < 16; ++j)
            a[j] = (__bf16)wrow[(ks << 5) + kmap(j, lane)];
#pragma unroll
        for (int nt = 0; nt < 4; ++nt) {      // 4 x 16 cols = head_dim 64
            v16bf bv;
#pragma unroll
            for (int j = 0; j < 16; ++j) {
                const int key = (ks << 5) + kmap(j, lane);
                bv[j] = Vp[((size_t)(b * SEQ + key)) * HID_ + h * HDIM +
                           (nt << 4) + (lane & 15)];
            }
            acc[nt] = __builtin_amdgcn_wmma_f32_16x16x32_bf16(
                false, a, false, bv, (short)0, acc[nt], false, false);
        }
    }
#pragma unroll
    for (int nt = 0; nt < 4; ++nt)
#pragma unroll
        for (int r = 0; r < 8; ++r) {
            const int m = q0 + r + ((lane >> 4) << 3);
            ctx[((size_t)(b * SEQ + m)) * HID_ + h * HDIM + (nt << 4) +
                (lane & 15)] = (__bf16)acc[nt][r];
        }
}

// ---------------------------------------------------------------------------
// Host-side launch
// ---------------------------------------------------------------------------
extern "C" void kernel_launch(void* const* d_in, const int* in_sizes, int n_in,
                              void* d_out, int out_size, void* d_ws, size_t ws_size,
                              hipStream_t stream) {
    const float* q   = (const float*)d_in[0];
    const float* k   = (const float*)d_in[1];
    const float* v   = (const float*)d_in[2];
    const float* Wq  = (const float*)d_in[3];
    const float* bq  = (const float*)d_in[4];
    const float* Wk  = (const float*)d_in[5];
    const float* bk  = (const float*)d_in[6];
    const float* Wv  = (const float*)d_in[7];
    const float* bv  = (const float*)d_in[8];
    const float* Wo  = (const float*)d_in[9];
    const float* bo  = (const float*)d_in[10];

    float* out   = (float*)d_out;                          // [B,T,HID]
    float* attnW = out + (size_t)BATCH * SEQ * HID_;       // [B,NH,T,T]

    const size_t IN  = (size_t)BATCH * SEQ * HID_;   // 8,388,608
    const size_t WEL = (size_t)HID_ * HID_;          // 1,048,576

    __bf16* ws = (__bf16*)d_ws;                      // ~126 MB of bf16 scratch
    __bf16* qb  = ws;                                // inputs (bf16)
    __bf16* kb  = qb + IN;
    __bf16* vb  = kb + IN;
    __bf16* wqb = vb + IN;                           // weights (bf16)
    __bf16* wkb = wqb + WEL;
    __bf16* wvb = wkb + WEL;
    __bf16* wob = wvb + WEL;
    __bf16* Qp  = wob + WEL;                         // projected Q/K/V (bf16)
    __bf16* Kp  = Qp + IN;
    __bf16* Vp  = Kp + IN;
    __bf16* ctx = Vp + IN;                           // attention context (bf16)

    // 0) f32 -> bf16 conversions
    {
        const int blk = 256;
        mha_cvt_f32_bf16<<<(int)((IN + blk - 1) / blk), blk, 0, stream>>>(q, qb, (int)IN);
        mha_cvt_f32_bf16<<<(int)((IN + blk - 1) / blk), blk, 0, stream>>>(k, kb, (int)IN);
        mha_cvt_f32_bf16<<<(int)((IN + blk - 1) / blk), blk, 0, stream>>>(v, vb, (int)IN);
        mha_cvt_f32_bf16<<<(int)((WEL + blk - 1) / blk), blk, 0, stream>>>(Wq, wqb, (int)WEL);
        mha_cvt_f32_bf16<<<(int)((WEL + blk - 1) / blk), blk, 0, stream>>>(Wk, wkb, (int)WEL);
        mha_cvt_f32_bf16<<<(int)((WEL + blk - 1) / blk), blk, 0, stream>>>(Wv, wvb, (int)WEL);
        mha_cvt_f32_bf16<<<(int)((WEL + blk - 1) / blk), blk, 0, stream>>>(Wo, wob, (int)WEL);
    }

    const int M = BATCH * SEQ;      // 8192
    const int N = HID_;             // 1024
    const int K = HID_;             // 1024
    dim3 ggrid(N / 128, M / 16);    // (8, 512)

    // 1) Q/K/V projections (bf16 out)
    mha_gemm_bf16_wmma<false><<<ggrid, 256, 0, stream>>>(qb, wqb, bq, nullptr, Qp, M, N, K);
    mha_gemm_bf16_wmma<false><<<ggrid, 256, 0, stream>>>(kb, wkb, bk, nullptr, Kp, M, N, K);
    mha_gemm_bf16_wmma<false><<<ggrid, 256, 0, stream>>>(vb, wvb, bv, nullptr, Vp, M, N, K);

    // 2) scores + softmax -> attention weights (f32, second output region)
    const int nblk = BATCH * NHEAD * (SEQ / 16);    // 8192
    mha_scores_softmax<<<nblk, 128, 0, stream>>>(Qp, Kp, attnW);

    // 3) context = attnW @ V
    mha_context<<<nblk, 32, 0, stream>>>(attnW, Vp, ctx);

    // 4) output projection (f32 out -> first output region)
    mha_gemm_bf16_wmma<true><<<ggrid, 256, 0, stream>>>(ctx, wob, bo, out, nullptr, M, N, K);
}
